// Double_SSM_Block_Custom_Channel_26285199852080
// MI455X (gfx1250) — matrix operand order repaired and checked by reading, not compiled, not run on hardware
//
#include <hip/hip_runtime.h>
#include <hip/hip_bf16.h>
#include <math.h>

typedef __attribute__((ext_vector_type(16))) _Float16 v16h;
typedef __attribute__((ext_vector_type(8)))  float    v8f;

#define L_SEQ  4096
#define NTOK   32768      // 8 * 4096
#define DIN    128
#define DBL_W  36

// ---------------------------------------------------------------------------
// K1: in_proj GEMM  xz[t, 0..255] = sum_c x[t,c] * W[j,c]
//   A: gathered from x1 (b, c, 4096)  -> row t = b*4096 + l, A[t][c] = x1[b][c][l]
//   M=32768, N=256, K=64. One wave per 16x16 tile, 2 WMMA k-steps.
// ---------------------------------------------------------------------------
__global__ void k_in_proj(const float* __restrict__ x1,
                          const float* __restrict__ w,
                          float* __restrict__ xz) {
  const int lane = threadIdx.x & 31;
  const int wid  = threadIdx.x >> 5;
  const int tile = blockIdx.x * 8 + wid;   // 32768 tiles total
  const int nt = tile & 15;                // N/16 = 16
  const int mt = tile >> 4;
  const int m0 = mt << 4, n0 = nt << 4;

  const int row = m0 + (lane & 15);
  const int b   = row >> 12;
  const int l   = row & 4095;
  const float* xbase = x1 + ((size_t)b << 18) + l;   // + c*4096
  const int col = n0 + (lane & 15);
  const float* wrow = w + (size_t)col * 64;

  const int kaOff = (lane & 16) ? 8  : 0;
  const int kbOff = (lane & 16) ? 8  : 0;

  v8f c = {0.f,0.f,0.f,0.f,0.f,0.f,0.f,0.f};
  #pragma unroll
  for (int k0 = 0; k0 < 64; k0 += 32) {
    v16h a, bf;
    #pragma unroll
    for (int e = 0; e < 16; ++e) {
      const int kk = k0 + kaOff + (e & 7) + ((e & 8) << 1);   // A-frag K map
      a[e] = (_Float16)xbase[(size_t)kk << 12];
    }
    #pragma unroll
    for (int e = 0; e < 16; ++e)
      bf[e] = (_Float16)wrow[k0 + kbOff + (e & 7) + ((e & 8) << 1)];   // B-frag K map
    c = __builtin_amdgcn_wmma_f32_16x16x32_f16(false, a, false, bf,
                                               (short)0, c, false, false);
  }
  const int mbase = m0 + ((lane & 16) ? 8 : 0);
  #pragma unroll
  for (int r = 0; r < 8; ++r)
    xz[(size_t)(mbase + r) * 256 + col] = c[r];
}

// ---------------------------------------------------------------------------
// K2: depthwise causal conv(4) + bias + SiLU on xs = xz[:, :128]
// ---------------------------------------------------------------------------
__global__ void k_conv_silu(const float* __restrict__ xz,
                            const float* __restrict__ cw,
                            const float* __restrict__ cb,
                            float* __restrict__ xs) {
  const int idx = blockIdx.x * blockDim.x + threadIdx.x;   // NTOK*128
  const int t = idx >> 7, d = idx & 127;
  const int l = t & 4095;
  float acc = cb[d];
  #pragma unroll
  for (int k = 0; k < 4; ++k) {
    const int ll = l - 3 + k;
    if (ll >= 0) acc += xz[(size_t)(t - 3 + k) * 256 + d] * cw[d * 4 + k];
  }
  xs[idx] = acc / (1.f + expf(-acc));   // SiLU
}

// ---------------------------------------------------------------------------
// K3: x_proj  dbl[t][j] = sum_d xs[t][d] * xpw[j][d],  j < 36
// ---------------------------------------------------------------------------
__global__ void k_x_proj(const float* __restrict__ xs,
                         const float* __restrict__ w,
                         float* __restrict__ dbl) {
  const int idx = blockIdx.x * blockDim.x + threadIdx.x;   // NTOK*36
  const int t = idx / DBL_W, j = idx % DBL_W;
  const float* xr = xs + (size_t)t * DIN;
  const float* wr = w  + (size_t)j * DIN;
  float acc = 0.f;
  #pragma unroll 4
  for (int d = 0; d < DIN; ++d) acc += xr[d] * wr[d];
  dbl[idx] = acc;
}

// ---------------------------------------------------------------------------
// K4: delta = softplus(dt @ dt_proj_w.T + dt_proj_b)
// ---------------------------------------------------------------------------
__global__ void k_delta(const float* __restrict__ dbl,
                        const float* __restrict__ dtw,
                        const float* __restrict__ dtb,
                        float* __restrict__ delta) {
  const int idx = blockIdx.x * blockDim.x + threadIdx.x;   // NTOK*128
  const int t = idx >> 7, d = idx & 127;
  const float* dr = dbl + (size_t)t * DBL_W;
  float acc = dtb[d];
  #pragma unroll
  for (int r = 0; r < 4; ++r) acc += dr[r] * dtw[d * 4 + r];
  delta[idx] = (acc > 20.f) ? acc : log1pf(expf(acc));
}

// ---------------------------------------------------------------------------
// K5: selective scan, fused with +xs*D and SiLU(z) gating.
//   Thread layout: n = tid&15 (state), dloc = tid>>4; the 16 n-threads of a
//   channel occupy one 16-lane half-wave -> shfl_xor reduction over n.
//   ys aliases delta (read-before-write within the same wave step).
// ---------------------------------------------------------------------------
__global__ void k_scan(const float* __restrict__ delta,
                       const float* __restrict__ xs,
                       const float* __restrict__ dbl,
                       const float* __restrict__ xz,
                       const float* __restrict__ A_log,
                       const float* __restrict__ Dvec,
                       float* __restrict__ ys) {
  const int tid  = threadIdx.x;
  const int n    = tid & 15;
  const int dloc = tid >> 4;
  const int batch = blockIdx.x >> 3;
  const int d = ((blockIdx.x & 7) << 4) + dloc;

  const float Aval = -expf(A_log[d * 16 + n]);
  const float Dd   = Dvec[d];
  float h = 0.f;
  const size_t tbase = (size_t)batch << 12;

  for (int l = 0; l < L_SEQ; ++l) {
    const size_t t  = tbase + l;
    const float dt = delta[t * 128 + d];
    const float xv = xs[t * 128 + d];
    const float Bv = dbl[t * DBL_W + 4 + n];
    const float Cv = dbl[t * DBL_W + 20 + n];
    h = expf(dt * Aval) * h + dt * Bv * xv;
    float yp = h * Cv;
    yp += __shfl_xor(yp, 1);
    yp += __shfl_xor(yp, 2);
    yp += __shfl_xor(yp, 4);
    yp += __shfl_xor(yp, 8);
    if (n == 0) {
      float yv = yp + xv * Dd;
      const float zv = xz[t * 256 + 128 + d];
      yv *= zv / (1.f + expf(-zv));          // * silu(z)
      ys[t * 128 + d] = yv;
    }
  }
}

// ---------------------------------------------------------------------------
// K6: out_proj GEMM  olin[t][c] = sum_d ys[t][d] * Wout[c][d]
//   M=32768, N=64, K=128. One wave per 16x16 tile, 4 WMMA k-steps.
// ---------------------------------------------------------------------------
__global__ void k_out_proj(const float* __restrict__ ys,
                           const float* __restrict__ w,
                           float* __restrict__ olin) {
  const int lane = threadIdx.x & 31;
  const int wid  = threadIdx.x >> 5;
  const int tile = blockIdx.x * 8 + wid;   // 8192 tiles
  const int nt = tile & 3;                 // N/16 = 4
  const int mt = tile >> 2;
  const int m0 = mt << 4, n0 = nt << 4;

  const int row = m0 + (lane & 15);
  const float* arow = ys + (size_t)row * DIN;
  const int col = n0 + (lane & 15);
  const float* wrow = w + (size_t)col * DIN;

  const int kaOff = (lane & 16) ? 8  : 0;
  const int kbOff = (lane & 16) ? 8  : 0;

  v8f c = {0.f,0.f,0.f,0.f,0.f,0.f,0.f,0.f};
  #pragma unroll
  for (int k0 = 0; k0 < DIN; k0 += 32) {
    v16h a, bf;
    #pragma unroll
    for (int e = 0; e < 16; ++e) {
      const int kk = k0 + kaOff + (e & 7) + ((e & 8) << 1);
      a[e] = (_Float16)arow[kk];
    }
    #pragma unroll
    for (int e = 0; e < 16; ++e)
      bf[e] = (_Float16)wrow[k0 + kbOff + (e & 7) + ((e & 8) << 1)];
    c = __builtin_amdgcn_wmma_f32_16x16x32_f16(false, a, false, bf,
                                               (short)0, c, false, false);
  }
  const int mbase = m0 + ((lane & 16) ? 8 : 0);
  #pragma unroll
  for (int r = 0; r < 8; ++r)
    olin[(size_t)(mbase + r) * 64 + col] = c[r];
}

// ---------------------------------------------------------------------------
// K7: LayerNorm(64) + NHWC->NCHW transpose.  One wave per token, 2 ch/lane.
// ---------------------------------------------------------------------------
__global__ void k_ln_out(const float* __restrict__ olin,
                         const float* __restrict__ g,
                         const float* __restrict__ bta,
                         float* __restrict__ out) {
  const int lane = threadIdx.x & 31;
  const int wid  = threadIdx.x >> 5;
  const int t = blockIdx.x * 4 + wid;
  const float v0 = olin[(size_t)t * 64 + lane];
  const float v1 = olin[(size_t)t * 64 + 32 + lane];
  float s = v0 + v1, q = v0 * v0 + v1 * v1;
  #pragma unroll
  for (int m = 1; m < 32; m <<= 1) {
    s += __shfl_xor(s, m);
    q += __shfl_xor(q, m);
  }
  const float mean = s * (1.f / 64.f);
  const float var  = q * (1.f / 64.f) - mean * mean;
  const float rstd = rsqrtf(var + 1e-5f);
  const int b = t >> 12, l = t & 4095;
  float* ob = out + ((size_t)b << 18) + l;     // out[b][c][l]
  ob[(size_t)lane << 12]        = (v0 - mean) * rstd * g[lane]      + bta[lane];
  ob[(size_t)(lane + 32) << 12] = (v1 - mean) * rstd * g[lane + 32] + bta[lane + 32];
}

// ---------------------------------------------------------------------------
extern "C" void kernel_launch(void* const* d_in, const int* in_sizes, int n_in,
                              void* d_out, int out_size, void* d_ws, size_t ws_size,
                              hipStream_t stream) {
  const float* x1        = (const float*)d_in[0];
  const float* in_proj_w = (const float*)d_in[1];
  const float* conv_w    = (const float*)d_in[2];
  const float* conv_b    = (const float*)d_in[3];
  const float* x_proj_w  = (const float*)d_in[4];
  const float* dt_proj_w = (const float*)d_in[5];
  const float* dt_proj_b = (const float*)d_in[6];
  const float* A_log     = (const float*)d_in[7];
  const float* Dvec      = (const float*)d_in[8];
  const float* out_proj_w= (const float*)d_in[9];
  const float* ln_g      = (const float*)d_in[10];
  const float* ln_b      = (const float*)d_in[11];
  float* out = (float*)d_out;

  char* ws = (char*)d_ws;
  float* xz    = (float*)(ws);                        // 32768*256*4 = 32 MB
  float* xs    = (float*)(ws + 33554432);             // 16 MB
  float* dbl   = (float*)(ws + 50331648);             // 32768*36*4 ~= 4.5 MB
  float* delta = (float*)(ws + 55050240);             // 16 MB   (total ~72 MB)
  float* ys    = delta;   // safe alias: read-before-write within same wave step
  float* olin  = xz;      // safe alias: z consumed by k_scan before k_out_proj

  k_in_proj  <<<4096,  256, 0, stream>>>(x1, in_proj_w, xz);
  k_conv_silu<<<16384, 256, 0, stream>>>(xz, conv_w, conv_b, xs);
  k_x_proj   <<<4608,  256, 0, stream>>>(xs, x_proj_w, dbl);
  k_delta    <<<16384, 256, 0, stream>>>(dbl, dt_proj_w, dt_proj_b, delta);
  k_scan     <<<64,    256, 0, stream>>>(delta, xs, dbl, xz, A_log, Dvec, ys);
  k_out_proj <<<1024,  256, 0, stream>>>(ys, out_proj_w, olin);
  k_ln_out   <<<8192,  128, 0, stream>>>(olin, ln_g, ln_b, out);
}
